// MultiHeadAttention_41721312313526
// MI455X (gfx1250) — matrix operand order repaired and checked by reading, not compiled, not run on hardware
//
#include <hip/hip_runtime.h>

// ---------------------------------------------------------------------------
// MultiHeadAttention with adjacency mask — MI455X (gfx1250, wave32, WMMA bf16)
// b=4, s=2048, e=512, h=8, d=64
// ---------------------------------------------------------------------------

#define Bn 4
#define Sn 2048
#define En 512
#define Hn 8
#define Dn 64
#define Mn (Bn * Sn)       // 8192
#define QKVN (3 * En)      // 1536

typedef __attribute__((ext_vector_type(16))) __bf16        v16bf;
typedef __attribute__((ext_vector_type(8)))  float         v8f;
typedef __attribute__((ext_vector_type(8)))  unsigned short v8u;

union Frag {
    v16bf v;
    v8u   h[2];
};

static __device__ __forceinline__ v8f zero8() {
    v8f z = {0.f, 0.f, 0.f, 0.f, 0.f, 0.f, 0.f, 0.f};
    return z;
}

// fp32 -> bf16 (round to nearest even; NaN ignored: inputs are well-scaled)
static __device__ __forceinline__ unsigned short f2bf(float f) {
    unsigned int u = __float_as_uint(f);
    u += 0x7FFFu + ((u >> 16) & 1u);
    return (unsigned short)(u >> 16);
}

static __device__ __forceinline__ v8f wmma_bf16(v16bf a, v16bf b, v8f c) {
    return __builtin_amdgcn_wmma_f32_16x16x32_bf16(
        /*neg_a=*/false, a, /*neg_b=*/false, b,
        /*c_mod=*/(short)0, c, /*reuse_a=*/false, /*reuse_b=*/false);
}

// A-matrix fragment (16x32 bf16): lane holds row (lane%16);
// elems 0..7  = K[kb + half*8 + 0..7], elems 8..15 = K[kb + 16 + half*8 + 0..7]
static __device__ __forceinline__ v16bf load_fragA(const unsigned short* base, int ld,
                                                   int row, int kb, int half) {
    Frag f;
    const unsigned short* p = base + row * ld + kb + half * 8;
    f.h[0] = *(const v8u*)(p);
    f.h[1] = *(const v8u*)(p + 16);
    return f.v;
}

// B-matrix fragment (32x16 bf16): lane holds column (lane%16);
// elems 0..7  = K[kb + half*8 + 0..7], elems 8..15 = K[kb + 16 + half*8 + 0..7]
static __device__ __forceinline__ v16bf load_fragB(const unsigned short* base, int ld,
                                                   int col, int kb, int half) {
    Frag f;
    const unsigned short* p = base + col * ld + kb + half * 8;
    f.h[0] = *(const v8u*)(p);
    f.h[1] = *(const v8u*)(p + 16);
    return f.v;
}

// CDNA5 async VMEM->LDS copy of 16 bytes (GLOBAL_LOAD_ASYNC_TO_LDS_B128).
// Per the flat-aperture rule, the low 32 bits of a generic LDS pointer are the
// LDS byte offset. Tracked by ASYNCcnt; drain with s_wait_asynccnt before use.
static __device__ __forceinline__ void async_copy_b128(const void* gsrc, void* ldst) {
    unsigned dst = (unsigned)(size_t)ldst;
    unsigned long long src = (unsigned long long)(size_t)gsrc;
    asm volatile("global_load_async_to_lds_b128 %0, %1, off"
                 :: "v"(dst), "v"(src) : "memory");
}
static __device__ __forceinline__ void async_wait0() {
    asm volatile("s_wait_asynccnt 0x0" ::: "memory");
}

#define LDP 72   // padded leading dim (elems): 16B aligned, bank-conflict free

// ---------------------------------------------------------------------------
// Kernel 1: qkv = x @ Wqkv^T + b ; scatter into Q/K/V (b,h,s,d) bf16,
// folding 1/sqrt(d) into Q.
// Block: 256 thr (8 waves, 4x2), block tile 128(M) x 64(N), K-step 64.
// ---------------------------------------------------------------------------
__global__ __launch_bounds__(256) void qkv_gemm_kernel(
    const float* __restrict__ x, const float* __restrict__ W,
    const float* __restrict__ bias,
    unsigned short* __restrict__ Qb, unsigned short* __restrict__ Kb,
    unsigned short* __restrict__ Vb) {
    __shared__ unsigned short sA[128 * LDP];
    __shared__ unsigned short sB[64 * LDP];

    const int tid  = threadIdx.x;
    const int wave = tid >> 5, lane = tid & 31;
    const int half = lane >> 4, l16 = lane & 15;
    const int wm = (wave & 3) * 32, wn = (wave >> 2) * 32;
    const int Mbase = blockIdx.x * 128, Nbase = blockIdx.y * 64;

    v8f acc[2][2];
#pragma unroll
    for (int i = 0; i < 2; ++i)
#pragma unroll
        for (int j = 0; j < 2; ++j) acc[i][j] = zero8();

    for (int k0 = 0; k0 < En; k0 += 64) {
        // stage A: 128x64 fp32 -> bf16 (2048 float4 chunks)
#pragma unroll
        for (int i = 0; i < 8; ++i) {
            int c = tid + i * 256;
            int row = c >> 4, kq = (c & 15) << 2;
            float4 f = *(const float4*)(x + (size_t)(Mbase + row) * En + k0 + kq);
            unsigned int lo = (unsigned int)f2bf(f.x) | ((unsigned int)f2bf(f.y) << 16);
            unsigned int hi = (unsigned int)f2bf(f.z) | ((unsigned int)f2bf(f.w) << 16);
            *(uint2*)(&sA[row * LDP + kq]) = make_uint2(lo, hi);
        }
        // stage B: 64x64 fp32 -> bf16 (W is (1536,512) row-major: W[n][k])
#pragma unroll
        for (int i = 0; i < 4; ++i) {
            int c = tid + i * 256;
            int row = c >> 4, kq = (c & 15) << 2;
            float4 f = *(const float4*)(W + (size_t)(Nbase + row) * En + k0 + kq);
            unsigned int lo = (unsigned int)f2bf(f.x) | ((unsigned int)f2bf(f.y) << 16);
            unsigned int hi = (unsigned int)f2bf(f.z) | ((unsigned int)f2bf(f.w) << 16);
            *(uint2*)(&sB[row * LDP + kq]) = make_uint2(lo, hi);
        }
        __syncthreads();

#pragma unroll
        for (int kk = 0; kk < 64; kk += 32) {
            v16bf a0 = load_fragA(sA, LDP, wm + l16,      kk, half);
            v16bf a1 = load_fragA(sA, LDP, wm + 16 + l16, kk, half);
            v16bf b0 = load_fragB(sB, LDP, wn + l16,      kk, half);
            v16bf b1 = load_fragB(sB, LDP, wn + 16 + l16, kk, half);
            acc[0][0] = wmma_bf16(a0, b0, acc[0][0]);
            acc[0][1] = wmma_bf16(a0, b1, acc[0][1]);
            acc[1][0] = wmma_bf16(a1, b0, acc[1][0]);
            acc[1][1] = wmma_bf16(a1, b1, acc[1][1]);
        }
        __syncthreads();
    }

    // epilogue: bias, scale Q by 1/sqrt(d), scatter to (b,h,s,d).
    // part/head derive from the lane-invariant tile base (16-col tile never
    // crosses a 512/64 boundary) -> pointer select, no exec divergence.
#pragma unroll
    for (int j = 0; j < 2; ++j) {
        int gnb  = Nbase + wn + j * 16;      // lane-invariant
        int part = gnb >> 9;                 // 0=q 1=k 2=v
        int eb   = gnb & 511;
        int head = eb >> 6;
        int db   = (eb & 63) + l16;
        unsigned short* dstp = (part == 0) ? Qb : (part == 1) ? Kb : Vb;
        float qs = (part == 0) ? 0.125f : 1.0f;
        float bv = bias[gnb + l16];
#pragma unroll
        for (int i = 0; i < 2; ++i) {
#pragma unroll
            for (int r = 0; r < 8; ++r) {
                int gm = Mbase + wm + i * 16 + r + half * 8;
                int bi = gm >> 11, si = gm & 2047;
                float v = (acc[i][j][r] + bv) * qs;
                size_t off = ((size_t)(bi * Hn + head) * Sn + si) * Dn + db;
                dstp[off] = f2bf(v);
            }
        }
    }
}

// ---------------------------------------------------------------------------
// Kernel 2: fused flash attention (per (b,h), 128 q rows per block, 16 per wave)
// K tile staged via async VMEM->LDS; V^T staged manually (transpose);
// adj mask + clip; online softmax with 16-lane shuffle reductions; P@V WMMA.
// ---------------------------------------------------------------------------
__global__ __launch_bounds__(256) void attn_flash_kernel(
    const unsigned short* __restrict__ Qb, const unsigned short* __restrict__ Kb,
    const unsigned short* __restrict__ Vb, const float* __restrict__ adj,
    unsigned short* __restrict__ Ob) {
    __shared__ unsigned short sK[64 * LDP];     // (k,d) row-major
    __shared__ unsigned short sVT[64 * LDP];    // (d,k) row-major (V transposed)
    __shared__ unsigned short sP[8 * 16 * LDP]; // per-wave P tile (16 x 64)

    const int tid  = threadIdx.x;
    const int wave = tid >> 5, lane = tid & 31;
    const int half = lane >> 4, l16 = lane & 15;

    const int bh = blockIdx.y;                // 0..31
    const int bi = bh >> 3;
    const unsigned short* Qp = Qb + (size_t)bh * Sn * Dn;
    const unsigned short* Kp = Kb + (size_t)bh * Sn * Dn;
    const unsigned short* Vp = Vb + (size_t)bh * Sn * Dn;
    const float* adjB = adj + (size_t)bi * Sn * Sn;

    const int qrow0 = blockIdx.x * 128 + wave * 16;
    unsigned short* sPw = &sP[wave * 16 * LDP];

    // Q fragments (d = 0..31 and 32..63), loaded once from global bf16
    v16bf aQ0 = load_fragA(Qp, Dn, qrow0 + l16, 0,  half);
    v16bf aQ1 = load_fragA(Qp, Dn, qrow0 + l16, 32, half);

    v8f O[4];
#pragma unroll
    for (int n = 0; n < 4; ++n) O[n] = zero8();
    float mrow[8], lrow[8];
#pragma unroll
    for (int r = 0; r < 8; ++r) { mrow[r] = -1.0e30f; lrow[r] = 0.f; }

    for (int k0 = 0; k0 < Sn; k0 += 64) {
        // stage K tile via CDNA5 async VMEM->LDS (2 x 16B per thread)
#pragma unroll
        for (int i = 0; i < 2; ++i) {
            int c = tid + i * 256;            // 512 chunks of 8 bf16
            int r = c >> 3, d8 = (c & 7) << 3;
            async_copy_b128(Kp + (size_t)(k0 + r) * Dn + d8, &sK[r * LDP + d8]);
            // V: load and transpose into LDS
            v8u vv = *(const v8u*)(Vp + (size_t)(k0 + r) * Dn + d8);
#pragma unroll
            for (int t = 0; t < 8; ++t) sVT[(d8 + t) * LDP + r] = vv[t];
        }
        async_wait0();
        __syncthreads();

        // scores: S = Q @ K^T (Q pre-scaled by 1/sqrt(d)); 4 tiles of 16 k-cols
        v8f Sc[4];
#pragma unroll
        for (int n = 0; n < 4; ++n) {
            v16bf bk0 = load_fragB(sK, LDP, n * 16 + l16, 0,  half);
            v16bf bk1 = load_fragB(sK, LDP, n * 16 + l16, 32, half);
            v8f s = wmma_bf16(aQ0, bk0, zero8());
            Sc[n]  = wmma_bf16(aQ1, bk1, s);
        }

        // adjacency mask, clip, block row-max
        float mb[8];
#pragma unroll
        for (int r = 0; r < 8; ++r) mb[r] = -1.0e30f;
#pragma unroll
        for (int n = 0; n < 4; ++n) {
            int kc = k0 + n * 16 + l16;
#pragma unroll
            for (int r = 0; r < 8; ++r) {
                int qr = qrow0 + r + half * 8;
                float a  = adjB[(size_t)qr * Sn + kc];
                float sv = Sc[n][r] * a;
                sv = fminf(fmaxf(sv, -10000.f), 10000.f);
                Sc[n][r] = sv;
                mb[r] = fmaxf(mb[r], sv);
            }
        }
#pragma unroll
        for (int r = 0; r < 8; ++r) {
#pragma unroll
            for (int msk = 1; msk < 16; msk <<= 1)
                mb[r] = fmaxf(mb[r], __shfl_xor(mb[r], msk, 16));
        }

        // online softmax update
        float scl[8], lsum[8];
#pragma unroll
        for (int r = 0; r < 8; ++r) {
            float mnew = fmaxf(mrow[r], mb[r]);
            scl[r]  = __expf(mrow[r] - mnew);
            mrow[r] = mnew;
            lsum[r] = 0.f;
        }
        // P = exp(S - m), accumulate row sums, write bf16 P to wave-private LDS
#pragma unroll
        for (int n = 0; n < 4; ++n) {
#pragma unroll
            for (int r = 0; r < 8; ++r) {
                float p = __expf(Sc[n][r] - mrow[r]);
                lsum[r] += p;
                sPw[(r + half * 8) * LDP + n * 16 + l16] = f2bf(p);
            }
        }
#pragma unroll
        for (int r = 0; r < 8; ++r) {
#pragma unroll
            for (int msk = 1; msk < 16; msk <<= 1)
                lsum[r] += __shfl_xor(lsum[r], msk, 16);
            lrow[r] = lrow[r] * scl[r] + lsum[r];
        }
        // rescale accumulator
#pragma unroll
        for (int n = 0; n < 4; ++n)
#pragma unroll
            for (int r = 0; r < 8; ++r) O[n][r] *= scl[r];

        // O += P @ V  (contraction over 64 k, via V^T tile)
        v16bf pa0 = load_fragA(sPw, LDP, l16, 0,  half);
        v16bf pa1 = load_fragA(sPw, LDP, l16, 32, half);
#pragma unroll
        for (int n = 0; n < 4; ++n) {
            v16bf bv0 = load_fragB(sVT, LDP, n * 16 + l16, 0,  half);
            v16bf bv1 = load_fragB(sVT, LDP, n * 16 + l16, 32, half);
            O[n] = wmma_bf16(pa0, bv0, O[n]);
            O[n] = wmma_bf16(pa1, bv1, O[n]);
        }

        // prefetch next adj block rows (gfx1250 global_prefetch_b8)
        if (k0 + 64 < Sn)
            __builtin_prefetch(adjB + (size_t)(qrow0 + l16) * Sn + k0 + 64, 0, 0);

        __syncthreads();
    }

    // normalize and write attention output as bf16 in (b, s, e) layout
    const int hh = bh & 7;
#pragma unroll
    for (int r = 0; r < 8; ++r) {
        int q = qrow0 + r + half * 8;
        float inv = 1.0f / lrow[r];
        size_t rowoff = ((size_t)bi * Sn + q) * En + hh * Dn;
#pragma unroll
        for (int n = 0; n < 4; ++n)
            Ob[rowoff + n * 16 + l16] = f2bf(O[n][r] * inv);
    }
}

// ---------------------------------------------------------------------------
// Kernel 3: out = attn @ out_w^T + out_b  (8192 x 512 x 512), fp32 result.
// A tile (already bf16) staged via async VMEM->LDS.
// ---------------------------------------------------------------------------
__global__ __launch_bounds__(256) void out_gemm_kernel(
    const unsigned short* __restrict__ A, const float* __restrict__ W,
    const float* __restrict__ bias, float* __restrict__ out) {
    __shared__ unsigned short sA[128 * LDP];
    __shared__ unsigned short sB[64 * LDP];

    const int tid  = threadIdx.x;
    const int wave = tid >> 5, lane = tid & 31;
    const int half = lane >> 4, l16 = lane & 15;
    const int wm = (wave & 3) * 32, wn = (wave >> 2) * 32;
    const int Mbase = blockIdx.x * 128, Nbase = blockIdx.y * 64;

    v8f acc[2][2];
#pragma unroll
    for (int i = 0; i < 2; ++i)
#pragma unroll
        for (int j = 0; j < 2; ++j) acc[i][j] = zero8();

    for (int k0 = 0; k0 < En; k0 += 64) {
        // stage A: 128x64 bf16 via async VMEM->LDS (1024 x 16B chunks)
#pragma unroll
        for (int i = 0; i < 4; ++i) {
            int c = tid + i * 256;
            int row = c >> 3, k8 = (c & 7) << 3;
            async_copy_b128(A + (size_t)(Mbase + row) * En + k0 + k8,
                            &sA[row * LDP + k8]);
        }
        // stage B: 64x64 fp32 -> bf16 (W is (512,512) row-major)
#pragma unroll
        for (int i = 0; i < 4; ++i) {
            int c = tid + i * 256;
            int row = c >> 4, kq = (c & 15) << 2;
            float4 f = *(const float4*)(W + (size_t)(Nbase + row) * En + k0 + kq);
            unsigned int lo = (unsigned int)f2bf(f.x) | ((unsigned int)f2bf(f.y) << 16);
            unsigned int hi = (unsigned int)f2bf(f.z) | ((unsigned int)f2bf(f.w) << 16);
            *(uint2*)(&sB[row * LDP + kq]) = make_uint2(lo, hi);
        }
        async_wait0();
        __syncthreads();

#pragma unroll
        for (int kk = 0; kk < 64; kk += 32) {
            v16bf a0 = load_fragA(sA, LDP, wm + l16,      kk, half);
            v16bf a1 = load_fragA(sA, LDP, wm + 16 + l16, kk, half);
            v16bf b0 = load_fragB(sB, LDP, wn + l16,      kk, half);
            v16bf b1 = load_fragB(sB, LDP, wn + 16 + l16, kk, half);
            acc[0][0] = wmma_bf16(a0, b0, acc[0][0]);
            acc[0][1] = wmma_bf16(a0, b1, acc[0][1]);
            acc[1][0] = wmma_bf16(a1, b0, acc[1][0]);
            acc[1][1] = wmma_bf16(a1, b1, acc[1][1]);
        }
        __syncthreads();
    }

#pragma unroll
    for (int i = 0; i < 2; ++i)
#pragma unroll
        for (int j = 0; j < 2; ++j) {
            int gn = Nbase + wn + j * 16 + l16;
            float bv = bias[gn];
#pragma unroll
            for (int r = 0; r < 8; ++r) {
                int gm = Mbase + wm + i * 16 + r + half * 8;
                out[(size_t)gm * En + gn] = acc[i][j][r] + bv;
            }
        }
}

// ---------------------------------------------------------------------------
extern "C" void kernel_launch(void* const* d_in, const int* in_sizes, int n_in,
                              void* d_out, int out_size, void* d_ws, size_t ws_size,
                              hipStream_t stream) {
    (void)in_sizes; (void)n_in; (void)out_size; (void)ws_size;
    const float* x      = (const float*)d_in[0];
    const float* adj    = (const float*)d_in[1];
    const float* Wqkv_w = (const float*)d_in[2];
    const float* Wqkv_b = (const float*)d_in[3];
    const float* out_w  = (const float*)d_in[4];
    const float* out_b  = (const float*)d_in[5];
    float* out = (float*)d_out;

    // workspace: Q | K | V | attn  (each 8 MiB of bf16)
    char* w = (char*)d_ws;
    unsigned short* Qb = (unsigned short*)(w);
    unsigned short* Kb = (unsigned short*)(w + (size_t)(8u << 20));
    unsigned short* Vb = (unsigned short*)(w + (size_t)(16u << 20));
    unsigned short* Ab = (unsigned short*)(w + (size_t)(24u << 20));

    qkv_gemm_kernel<<<dim3(Mn / 128, QKVN / 64), 256, 0, stream>>>(
        x, Wqkv_w, Wqkv_b, Qb, Kb, Vb);
    attn_flash_kernel<<<dim3(Sn / 128, Bn * Hn), 256, 0, stream>>>(
        Qb, Kb, Vb, adj, Ab);
    out_gemm_kernel<<<dim3(Mn / 128, En / 64), 256, 0, stream>>>(
        Ab, out_w, out_b, out);
}
